// Grid2MeshNodeUpdate_21998822490254
// MI455X (gfx1250) — compile-verified, matrix operand order checked by reading, not run on hardware
//
#include <hip/hip_runtime.h>

typedef __attribute__((ext_vector_type(16))) __bf16 v16bf;
typedef __attribute__((ext_vector_type(8)))  __bf16 v8bf;
typedef __attribute__((ext_vector_type(4)))  __bf16 v4bf;
typedef __attribute__((ext_vector_type(8)))  float  v8f;

// ---------------------------------------------------------------------------
// WMMA helpers (CDNA5 gfx1250, wave32)
// ---------------------------------------------------------------------------
__device__ __forceinline__ v8f wmma_bf16(v16bf a, v16bf b, v8f c) {
  // D = A(16x32 bf16) * B(32x16 bf16) + C(16x16 f32)
  return __builtin_amdgcn_wmma_f32_16x16x32_bf16(
      /*neg_a=*/false, a, /*neg_b=*/false, b,
      /*c_mod=*/(short)0, c, /*reuse_a=*/false, /*reuse_b=*/false);
}

// A fragment: lane L holds row (L%16); K = kt*32 + (L/16)*8 .. +7  and  +16..+23
__device__ __forceinline__ v16bf load_a_frag(const __bf16* A, int lda, int kt) {
  const int lane = threadIdx.x & 31;
  const __bf16* p = A + (lane & 15) * lda + kt * 32 + ((lane >> 4) << 3);
  union { v16bf v; v8bf h[2]; } u;
  u.h[0] = *(const v8bf*)(p);
  u.h[1] = *(const v8bf*)(p + 16);
  return u.v;
}

// B fragment: prepacked so each lane reads 16 contiguous bf16 (32B)
__device__ __forceinline__ v16bf load_b_frag(const __bf16* W, int tile) {
  const int lane = threadIdx.x & 31;
  return *(const v16bf*)(W + ((size_t)tile * 32 + lane) * 16);
}

__device__ __forceinline__ float silu_f(float x) {
  return x * (1.0f / (1.0f + __expf(-x)));
}

// Warm GL2 with a weight matrix (one prefetch per 256B cacheline region/lane)
__device__ __forceinline__ void prefetch_weights(const __bf16* W, int bytes) {
  const int t = threadIdx.x;
  const int stride = blockDim.x * 256;
  for (int o = t * 256; o < bytes; o += stride)
    __builtin_prefetch((const char*)W + o, 0, 1);   // -> global_prefetch_b8
}

// ---------------------------------------------------------------------------
// One wave computes a 16-row tile: O[16,N] = act(A[16,K] @ W + bias) -> LDS bf16
// KT = K/32, NT = N/16. N-tiles processed 4 at a time (32 accum VGPRs).
// ---------------------------------------------------------------------------
template<int KT, int NT, bool SILU>
__device__ __forceinline__ void gemm_to_lds(const __bf16* A, int lda,
                                            const __bf16* W, const float* bias,
                                            __bf16* O, int ldo) {
  const int lane  = threadIdx.x & 31;
  const int nloc  = lane & 15;
  const int mbase = (lane >> 4) << 3;
  for (int nc = 0; nc < NT; nc += 4) {
    v8f acc[4] = {};
    for (int kt = 0; kt < KT; ++kt) {
      v16bf a = load_a_frag(A, lda, kt);
#pragma unroll
      for (int j = 0; j < 4; ++j) {
        v16bf b = load_b_frag(W, kt * NT + nc + j);
        acc[j] = wmma_bf16(a, b, acc[j]);
      }
    }
#pragma unroll
    for (int j = 0; j < 4; ++j) {
      const int n0 = (nc + j) * 16;
      const float bv = bias[n0 + nloc];
#pragma unroll
      for (int p = 0; p < 8; ++p) {
        float v = acc[j][p] + bv;     // D element (m = mbase+p, n = n0+nloc)
        if (SILU) v = silu_f(v);
        O[(mbase + p) * ldo + n0 + nloc] = (__bf16)v;
      }
    }
  }
}

// Final layer: f32 output to global (pre-LayerNorm) + global sum/sumsq stats.
template<int KT, int NT>
__device__ __forceinline__ void gemm_final(const __bf16* A, int lda,
                                           const __bf16* W, const float* bias,
                                           float* out, int row0, int nrows,
                                           float* stats) {
  const int lane  = threadIdx.x & 31;
  const int nloc  = lane & 15;
  const int mbase = (lane >> 4) << 3;
  float s = 0.f, sq = 0.f;
  for (int nc = 0; nc < NT; nc += 4) {
    v8f acc[4] = {};
    for (int kt = 0; kt < KT; ++kt) {
      v16bf a = load_a_frag(A, lda, kt);
#pragma unroll
      for (int j = 0; j < 4; ++j) {
        v16bf b = load_b_frag(W, kt * NT + nc + j);
        acc[j] = wmma_bf16(a, b, acc[j]);
      }
    }
#pragma unroll
    for (int j = 0; j < 4; ++j) {
      const int n0 = (nc + j) * 16;
      const float bv = bias[n0 + nloc];
#pragma unroll
      for (int p = 0; p < 8; ++p) {
        const int gr = row0 + mbase + p;
        float v = acc[j][p] + bv;
        if (gr < nrows) {
          out[(size_t)gr * (NT * 16) + n0 + nloc] = v;
          s += v; sq += v * v;
        }
      }
    }
  }
  // wave32 reduction, one atomic pair per wave
  for (int off = 16; off > 0; off >>= 1) {
    s  += __shfl_down(s, off);
    sq += __shfl_down(sq, off);
  }
  if (lane == 0) { atomicAdd(&stats[0], s); atomicAdd(&stats[1], sq); }
}

// ---------------------------------------------------------------------------
// Kernels
// ---------------------------------------------------------------------------
__global__ void zero_ws_kernel(float4* __restrict__ eagg4,
                               float* __restrict__ stats, size_t n4) {
  size_t i = (size_t)blockIdx.x * blockDim.x + threadIdx.x;
  if (i < 4) stats[i] = 0.f;
  if (i < n4) eagg4[i] = make_float4(0.f, 0.f, 0.f, 0.f);
}

// Pack W[K,N] f32 row-major -> bf16 WMMA B-fragments (32x16 per tile).
// Packed element t: j=t&15, lane=(t>>4)&31, tile=t>>9; nt=tile%NT, kt=tile/NT.
__global__ void pack_kernel(const float* __restrict__ W, __bf16* __restrict__ P,
                            int K, int N) {
  int t = blockIdx.x * blockDim.x + threadIdx.x;
  if (t >= K * N) return;
  const int j    = t & 15;
  const int lane = (t >> 4) & 31;
  const int tile = t >> 9;
  const int NT   = N >> 4;
  const int nt   = tile % NT;
  const int kt   = tile / NT;
  const int n = nt * 16 + (lane & 15);
  const int k = kt * 32 + ((lane >> 4) << 3) + j + ((j >= 8) ? 8 : 0);
  P[t] = (__bf16)W[(size_t)k * N + n];
}

// 4 columns per thread: 1 float4 load + 1 index load + 4 f32 atomics.
__global__ void scatter_kernel(const float4* __restrict__ ex4,
                               const int* __restrict__ col,
                               float* __restrict__ eagg, long total4) {
  long i = (long)blockIdx.x * blockDim.x + threadIdx.x;
  if (i >= total4) return;
  const int e  = (int)(i >> 5);
  const int c4 = (int)(i & 31) << 2;
  const float4 v = ex4[i];
  float* dst = &eagg[(size_t)col[e] * 128 + c4];
  atomicAdd(dst + 0, v.x);
  atomicAdd(dst + 1, v.y);
  atomicAdd(dst + 2, v.z);
  atomicAdd(dst + 3, v.w);
}

// Mesh: [M,256] -> 512 (SiLU) -> 256 (SiLU) -> 128; 2 waves/block, 16 rows/wave.
__global__ __launch_bounds__(64) void mesh_mlp_kernel(
    const float* __restrict__ mx, const float* __restrict__ eagg,
    const __bf16* __restrict__ pw1, const float* __restrict__ b1,
    const __bf16* __restrict__ pw2, const float* __restrict__ b2,
    const __bf16* __restrict__ pw3, const float* __restrict__ b3,
    float* __restrict__ hpre, float* __restrict__ stats, int nrows) {
  constexpr int LDA = 256 + 8;   // bf16 elems, +8 pad breaks bank conflicts
  constexpr int LDY = 512 + 8;
  __shared__ __align__(16) __bf16 sA[2][16 * LDA];  // input, reused for y2
  __shared__ __align__(16) __bf16 sY[2][16 * LDY];  // y1
  const int wave = threadIdx.x >> 5;
  const int lane = threadIdx.x & 31;
  const int row0 = (blockIdx.x * 2 + wave) * 16;
  __bf16* A = sA[wave];
  __bf16* Y = sY[wave];
  if (blockIdx.x == 0) prefetch_weights(pw1, 256 * 512 * 2);
  // stage concat([mx, edge_agg]) as bf16: float4 loads, b64 LDS stores
  for (int i = lane * 4; i < 16 * 256; i += 32 * 4) {
    const int r = i >> 8, c = i & 255;
    const int gr = row0 + r;
    float4 v = make_float4(0.f, 0.f, 0.f, 0.f);
    if (gr < nrows) {
      const float* src = (c < 128) ? &mx[(size_t)gr * 128 + c]
                                   : &eagg[(size_t)gr * 128 + (c - 128)];
      v = *(const float4*)src;
    }
    v4bf o = { (__bf16)v.x, (__bf16)v.y, (__bf16)v.z, (__bf16)v.w };
    *(v4bf*)(&A[r * LDA + c]) = o;
  }
  __syncthreads();
  gemm_to_lds<8, 32, true>(A, LDA, pw1, b1, Y, LDY);    // 256 -> 512, SiLU
  __syncthreads();
  gemm_to_lds<16, 16, true>(Y, LDY, pw2, b2, A, LDA);   // 512 -> 256, SiLU
  __syncthreads();
  gemm_final<8, 8>(A, LDA, pw3, b3, hpre, row0, nrows, stats);  // 256 -> 128
}

// Grid: [G,128] -> 256 (SiLU) -> 128; 4 waves/block, 16 rows/wave.
__global__ __launch_bounds__(128) void grid_mlp_kernel(
    const float* __restrict__ gx,
    const __bf16* __restrict__ pw1, const float* __restrict__ b1,
    const __bf16* __restrict__ pw2, const float* __restrict__ b2,
    float* __restrict__ gpre, float* __restrict__ stats, int nrows) {
  constexpr int LDA = 128 + 8;
  constexpr int LDY = 256 + 8;
  __shared__ __align__(16) __bf16 sA[4][16 * LDA];
  __shared__ __align__(16) __bf16 sY[4][16 * LDY];
  const int wave = threadIdx.x >> 5;
  const int lane = threadIdx.x & 31;
  const int row0 = (blockIdx.x * 4 + wave) * 16;
  __bf16* A = sA[wave];
  __bf16* Y = sY[wave];
  if (blockIdx.x == 0) prefetch_weights(pw1, 128 * 256 * 2);
  for (int i = lane * 4; i < 16 * 128; i += 32 * 4) {
    const int r = i >> 7, c = i & 127;
    const int gr = row0 + r;
    float4 v = make_float4(0.f, 0.f, 0.f, 0.f);
    if (gr < nrows) v = *(const float4*)&gx[(size_t)gr * 128 + c];
    v4bf o = { (__bf16)v.x, (__bf16)v.y, (__bf16)v.z, (__bf16)v.w };
    *(v4bf*)(&A[r * LDA + c]) = o;
  }
  __syncthreads();
  gemm_to_lds<4, 16, true>(A, LDA, pw1, b1, Y, LDY);    // 128 -> 256, SiLU
  __syncthreads();
  gemm_final<8, 8>(Y, LDY, pw2, b2, gpre, row0, nrows, stats);  // 256 -> 128
}

// Fused: io = base + (io - mu)*rsqrt(var+eps)*lnw + lnb   (full-tensor LN)
// float4-vectorized; count4 = count/4 (feature dim 128 => always divisible).
__global__ void finalize_kernel(const float4* __restrict__ base,
                                const float4* __restrict__ lnw,
                                const float4* __restrict__ lnb,
                                float4* __restrict__ io,
                                const float* __restrict__ stats,
                                float invn, size_t count4) {
  size_t i = (size_t)blockIdx.x * blockDim.x + threadIdx.x;
  if (i >= count4) return;
  const float mu  = stats[0] * invn;
  const float var = stats[1] * invn - mu * mu;
  const float rs  = rsqrtf(var + 1e-5f);
  const float4 v = io[i];
  const float4 b = base[i];
  const float4 w = lnw[i];
  const float4 o = lnb[i];
  float4 r;
  r.x = b.x + (v.x - mu) * rs * w.x + o.x;
  r.y = b.y + (v.y - mu) * rs * w.y + o.y;
  r.z = b.z + (v.z - mu) * rs * w.z + o.z;
  r.w = b.w + (v.w - mu) * rs * w.w + o.w;
  io[i] = r;
}

// ---------------------------------------------------------------------------
// Host launch
// ---------------------------------------------------------------------------
extern "C" void kernel_launch(void* const* d_in, const int* in_sizes, int n_in,
                              void* d_out, int out_size, void* d_ws, size_t ws_size,
                              hipStream_t stream) {
  const float* gx      = (const float*)d_in[0];
  const float* mx      = (const float*)d_in[1];
  const int*   g2me_i  = (const int*)  d_in[4];
  const float* g2me_x  = (const float*)d_in[5];
  const float* agg_w1  = (const float*)d_in[8];
  const float* agg_b1  = (const float*)d_in[9];
  const float* agg_w2  = (const float*)d_in[10];
  const float* agg_b2  = (const float*)d_in[11];
  const float* agg_w3  = (const float*)d_in[12];
  const float* agg_b3  = (const float*)d_in[13];
  const float* agg_ln_w = (const float*)d_in[14];
  const float* agg_ln_b = (const float*)d_in[15];
  const float* grid_w1 = (const float*)d_in[16];
  const float* grid_b1 = (const float*)d_in[17];
  const float* grid_w2 = (const float*)d_in[18];
  const float* grid_b2 = (const float*)d_in[19];
  const float* grid_ln_w = (const float*)d_in[20];
  const float* grid_ln_b = (const float*)d_in[21];

  const int GN = in_sizes[0] / 128;   // 262144 grid nodes
  const int MN = in_sizes[1] / 128;   // 40962 mesh nodes
  const int E  = in_sizes[4] / 2;     // 600000 grid2mesh edges

  float* outf = (float*)d_out;
  size_t off[8]; off[0] = 0;
  for (int i = 1; i < 8; ++i) off[i] = off[i - 1] + (size_t)in_sizes[i - 1];

  // Workspace layout (byte offsets, 256-aligned)
  char*  ws    = (char*)d_ws;
  float* stats = (float*)ws;                      // [4]: meshSum,meshSq,gridSum,gridSq
  float* eagg  = (float*)(ws + 256);              // [MN,128] f32
  size_t eaggBytes = (size_t)MN * 128 * sizeof(float);
  size_t pwOff = 256 + ((eaggBytes + 255) & ~(size_t)255);
  __bf16* pw1 = (__bf16*)(ws + pwOff);            // 256x512 bf16 packed
  __bf16* pw2 = (__bf16*)(ws + pwOff + 262144);   // 512x256
  __bf16* pw3 = (__bf16*)(ws + pwOff + 524288);   // 256x128
  __bf16* pg1 = (__bf16*)(ws + pwOff + 589824);   // 128x256
  __bf16* pg2 = (__bf16*)(ws + pwOff + 655360);   // 256x128

  // 1) zero edge aggregate + LN stats (float4)
  {
    size_t n4 = (size_t)MN * 128 / 4;
    zero_ws_kernel<<<(int)((n4 + 255) / 256), 256, 0, stream>>>(
        (float4*)eagg, stats, n4);
  }
  // 2) pack weights to bf16 B-fragment layout (tiny; re-done every launch)
  pack_kernel<<<(256 * 512 + 255) / 256, 256, 0, stream>>>(agg_w1, pw1, 256, 512);
  pack_kernel<<<(512 * 256 + 255) / 256, 256, 0, stream>>>(agg_w2, pw2, 512, 256);
  pack_kernel<<<(256 * 128 + 255) / 256, 256, 0, stream>>>(agg_w3, pw3, 256, 128);
  pack_kernel<<<(128 * 256 + 255) / 256, 256, 0, stream>>>(grid_w1, pg1, 128, 256);
  pack_kernel<<<(256 * 128 + 255) / 256, 256, 0, stream>>>(grid_w2, pg2, 256, 128);
  // 3) scatter-sum edges onto mesh nodes (f32 atomics, float4 loads)
  {
    long total4 = (long)E * 32;
    scatter_kernel<<<(int)((total4 + 255) / 256), 256, 0, stream>>>(
        (const float4*)g2me_x, g2me_i + E, eagg, total4);
  }
  // 4) mesh MLP (pre-LN result lands in d_out mx region)
  mesh_mlp_kernel<<<(MN + 31) / 32, 64, 0, stream>>>(
      mx, eagg, pw1, agg_b1, pw2, agg_b2, pw3, agg_b3,
      outf + off[1], stats, MN);
  // 5) grid MLP (pre-LN result lands in d_out gx region)
  grid_mlp_kernel<<<(GN + 63) / 64, 128, 0, stream>>>(
      gx, pg1, grid_b1, pg2, grid_b2,
      outf + off[0], stats + 2, GN);
  // 6) fused residual + full-tensor LayerNorm (in-place on d_out, float4)
  {
    size_t cm = (size_t)MN * 128;
    finalize_kernel<<<(int)((cm / 4 + 255) / 256), 256, 0, stream>>>(
        (const float4*)mx, (const float4*)agg_ln_w, (const float4*)agg_ln_b,
        (float4*)(outf + off[1]), stats, 1.0f / (float)cm, cm / 4);
    size_t cg = (size_t)GN * 128;
    finalize_kernel<<<(int)((cg / 4 + 255) / 256), 256, 0, stream>>>(
        (const float4*)gx, (const float4*)grid_ln_w, (const float4*)grid_ln_b,
        (float4*)(outf + off[0]), stats + 2, 1.0f / (float)cg, cg / 4);
  }
  // 7) pass-through outputs (capture-safe async D2D copies)
  for (int k = 2; k < 8; ++k) {
    hipMemcpyAsync(outf + off[k], d_in[k], (size_t)in_sizes[k] * sizeof(float),
                   hipMemcpyDeviceToDevice, stream);
  }
}
